// UELM4_50233937494165
// MI455X (gfx1250) — matrix operand order repaired and checked by reading, not compiled, not run on hardware
//
#include <hip/hip_runtime.h>
#include <hip/hip_bf16.h>
#include <math.h>

#define D_DIM   1024
#define VOCAB   32000
#define KMEM    32768
#define NTOK    4096
#define SHORT_K 32
#define BANDW   127
#define CHUNK_M 512

typedef __attribute__((ext_vector_type(16))) __bf16 v16bf;
typedef __attribute__((ext_vector_type(8)))  float  v8f;
typedef __attribute__((ext_vector_type(2)))  float  v2f;

union BF16x16 { uint4 q[2]; v16bf v; };

__device__ __forceinline__ unsigned short f32_bf16(float f) {
  unsigned int u = __float_as_uint(f);
  u += 0x7FFFu + ((u >> 16) & 1u);           // round-to-nearest-even
  return (unsigned short)(u >> 16);
}

// ---------------- elementwise / prep kernels ----------------

__global__ void k_cvt_bf16(const float* __restrict__ in, unsigned short* __restrict__ out, long n) {
  long i = (long)blockIdx.x * blockDim.x + threadIdx.x;
  long stride = (long)gridDim.x * blockDim.x;
  for (; i < n; i += stride) out[i] = f32_bf16(in[i]);
}

__global__ void k_zero(float* __restrict__ p, long n) {
  long i = (long)blockIdx.x * blockDim.x + threadIdx.x;
  long stride = (long)gridDim.x * blockDim.x;
  for (; i < n; i += stride) p[i] = 0.0f;
}

// out[C,R] (bf16) = transpose of in[R,C] (f32)
__global__ void k_transpose_bf16(const float* __restrict__ in, unsigned short* __restrict__ out,
                                 int R, int C) {
  __shared__ float tile[16][17];
  int c0 = blockIdx.x * 16, r0 = blockIdx.y * 16;
  int tx = threadIdx.x, ty = threadIdx.y;
  tile[ty][tx] = in[(size_t)(r0 + ty) * C + (c0 + tx)];
  __syncthreads();
  out[(size_t)(c0 + ty) * R + (r0 + tx)] = f32_bf16(tile[tx][ty]);
}

__global__ void k_embed_gather(const int* __restrict__ tokens, const float* __restrict__ embed,
                               float* __restrict__ Ef, unsigned short* __restrict__ Eb) {
  int t = blockIdx.x;
  int row = tokens[t];
  const float* src = embed + (size_t)row * D_DIM;
  for (int d = threadIdx.x; d < D_DIM; d += blockDim.x) {
    float v = src[d];
    Ef[(size_t)t * D_DIM + d] = v;
    Eb[(size_t)t * D_DIM + d] = f32_bf16(v);
  }
}

// A_m[e,d] = field_W[e,d] * (|e-d|<=BAND); AT_m = transpose
__global__ void k_band(const float* __restrict__ fw, float* __restrict__ Am, float* __restrict__ ATm) {
  int i = blockIdx.x * blockDim.x + threadIdx.x;
  if (i >= D_DIM * D_DIM) return;
  int e = i / D_DIM, d = i - e * D_DIM;
  int diff = e - d; if (diff < 0) diff = -diff;
  float v = (diff <= BANDW) ? fw[i] : 0.0f;
  Am[i] = v;
  ATm[(size_t)d * D_DIM + e] = v;
}

// ---------------- bf16 NT GEMM: C[M,N] = A[M,K] * B[N,K]^T (+bias over N) ----------------
// Block: 256 threads = 8 waves, block tile 64(M) x 64(N).
// B panel (64 cols x 32 k, bf16) staged in LDS with async global->LDS copies
// (double-buffered, ASYNCcnt); each wave: 1 M-tile x 2 N-tiles, A frag reused.
// Requires M%64==0, N%64==0, K%32==0 (always true for our shapes).

__global__ void __launch_bounds__(256) k_gemm_nt_bf16(
    const unsigned short* __restrict__ A,
    const unsigned short* __restrict__ B,
    float* __restrict__ C,
    const float* __restrict__ bias,
    int M, int N, int K) {
  __shared__ unsigned short Bs[2][64 * 32];

  int lane = threadIdx.x & 31;
  int wid  = threadIdx.x >> 5;
  int wm = wid & 3;                 // 4 m-tiles
  int wn = wid >> 2;                // 2 n-groups of 32 cols
  int m0   = blockIdx.y * 64 + wm * 16;
  int nblk = blockIdx.x * 64;
  bool hi = lane >= 16;

  const unsigned short* Ar = A + (size_t)(m0 + (lane & 15)) * K;

  // cooperative async B-panel copy: thread t moves 16B (8 bf16)
  int ccol = threadIdx.x >> 2;            // 0..63
  int cko  = (threadIdx.x & 3) * 8;       // 0,8,16,24
  const unsigned short* Bg = B + (size_t)(nblk + ccol) * K + cko;
  unsigned lds0 = (unsigned)(size_t)&Bs[0][ccol * 32 + cko];
  unsigned lds1 = (unsigned)(size_t)&Bs[1][ccol * 32 + cko];

  #define ISSUE_B(bufaddr, k0)                                              \
    asm volatile("global_load_async_to_lds_b128 %0, %1, off"                \
                 :: "v"(bufaddr), "v"(Bg + (k0)) : "memory")

  v8f acc0 = {}, acc1 = {};
  ISSUE_B(lds0, 0);

  int KT = K >> 5;
  for (int kt = 0; kt < KT; ++kt) {
    int cur = kt & 1;
    asm volatile("s_wait_asynccnt 0x0" ::: "memory");
    __syncthreads();                               // B panel [cur] visible to all
    if (kt + 1 < KT) {
      if (cur) ISSUE_B(lds0, (kt + 1) << 5);
      else     ISSUE_B(lds1, (kt + 1) << 5);
    }

    int k0 = kt << 5;
    BF16x16 a, b0, b1;
    int ka = k0 + (hi ? 16 : 0);                   // A 16x32 bf16 lane layout
    a.q[0] = *(const uint4*)(Ar + ka);
    a.q[1] = *(const uint4*)(Ar + ka + 8);
    if (k0 + 128 < K) __builtin_prefetch(Ar + k0 + 128, 0, 3);

    int koff = hi ? 16 : 0;                        // B 32x16 bf16 lane layout
    int c0 = wn * 32 + (lane & 15);
    const unsigned short* bp0 = &Bs[cur][c0 * 32 + koff];
    const unsigned short* bp1 = &Bs[cur][(c0 + 16) * 32 + koff];
    b0.q[0] = *(const uint4*)bp0;  b0.q[1] = *(const uint4*)(bp0 + 8);
    b1.q[0] = *(const uint4*)bp1;  b1.q[1] = *(const uint4*)(bp1 + 8);

    acc0 = __builtin_amdgcn_wmma_f32_16x16x32_bf16(false, a.v, false, b0.v,
                                                   (short)0, acc0, false, false);
    acc1 = __builtin_amdgcn_wmma_f32_16x16x32_bf16(false, a.v, false, b1.v,
                                                   (short)0, acc1, false, false);
  }
  #undef ISSUE_B

  int crow = m0 + (hi ? 8 : 0);
  int col0 = nblk + wn * 32 + (lane & 15);
  float bv0 = bias ? bias[col0]      : 0.0f;
  float bv1 = bias ? bias[col0 + 16] : 0.0f;
  #pragma unroll
  for (int r = 0; r < 8; ++r) {
    C[(size_t)(crow + r) * N + col0]      = acc0[r] + bv0;
    C[(size_t)(crow + r) * N + col0 + 16] = acc1[r] + bv1;
  }
}

// ---------------- f32 NT GEMM (16x16x4 WMMA), optional band-limited K range ----------------

__global__ void k_gemm_nt_f32(const float* __restrict__ A,
                              const float* __restrict__ B,
                              float* __restrict__ C,
                              int M, int N, int K, int banded) {
  int lane = threadIdx.x & 31;
  int wid  = threadIdx.x >> 5;
  int wm = wid & 3, wn = wid >> 2;
  int m0 = (blockIdx.y * 4 + wm) * 16;
  int n0 = (blockIdx.x * 2 + wn) * 16;
  if (m0 >= M || n0 >= N) return;
  bool hi = lane >= 16;
  const float* Ar = A + (size_t)(m0 + (lane & 15)) * K;
  const float* Br = B + (size_t)(n0 + (lane & 15)) * K;
  int klo = 0, khi = K;
  if (banded) {
    klo = n0 - BANDW; if (klo < 0) klo = 0; klo &= ~7;
    khi = n0 + 16 + BANDW; khi = (khi + 7) & ~7; if (khi > K) khi = K;
  }
  v8f acc = {};
  int off = hi ? 2 : 0;             // f32 A 16x4: lanes>=16 hold K=2,3
  for (int k0 = klo; k0 < khi; k0 += 8) {
    float2 av0 = *(const float2*)(Ar + k0 + off);
    float2 bv0 = *(const float2*)(Br + k0 + off);
    float2 av1 = *(const float2*)(Ar + k0 + 4 + off);
    float2 bv1 = *(const float2*)(Br + k0 + 4 + off);
    v2f a0; a0[0] = av0.x; a0[1] = av0.y;
    v2f b0; b0[0] = bv0.x; b0[1] = bv0.y;
    acc = __builtin_amdgcn_wmma_f32_16x16x4_f32(false, a0, false, b0,
                                                (short)0, acc, false, false);
    v2f a1; a1[0] = av1.x; a1[1] = av1.y;
    v2f b1; b1[0] = bv1.x; b1[1] = bv1.y;
    acc = __builtin_amdgcn_wmma_f32_16x16x4_f32(false, a1, false, b1,
                                                (short)0, acc, false, false);
  }
  int crow = m0 + (hi ? 8 : 0);
  int ccol = n0 + (lane & 15);
  #pragma unroll
  for (int r = 0; r < 8; ++r)
    C[(size_t)(crow + r) * N + ccol] = acc[r];
}

// ---------------- top-32 of 32768 per token: 1 wave/token ----------------

__global__ void k_topk(const float* __restrict__ scores, int* __restrict__ kset, int tok0) {
  __shared__ float sv[32 * 32];
  __shared__ int   si[32 * 32];
  int lane = threadIdx.x;
  int t = blockIdx.x;
  const float* row = scores + (size_t)t * KMEM;
  int base = lane * 32;
  for (int j = 0; j < 32; ++j) { sv[base + j] = -3.4e38f; si[base + j] = -1; }
  // per-lane top-32 over a disjoint strided slice (indices unique per lane)
  for (int i = lane; i < KMEM; i += 32) {
    float v = row[i];
    if (v > sv[base]) {
      int p = 0;
      while (p < 31 && sv[base + p + 1] < v) {
        sv[base + p] = sv[base + p + 1];
        si[base + p] = si[base + p + 1];
        ++p;
      }
      sv[base + p] = v; si[base + p] = i;
    }
  }
  // 32 rounds of cross-lane max-pop
  int pos = 31;
  int* out = kset + (size_t)(tok0 + t) * SHORT_K;
  for (int r = 0; r < SHORT_K; ++r) {
    float cv = (pos >= 0) ? sv[base + pos] : -3.4e38f;
    int   ci = (pos >= 0) ? si[base + pos] : -1;
    float bv = cv; int bi = ci;
    for (int off = 16; off; off >>= 1) {
      float ov = __shfl_xor(bv, off, 32);
      int   oi = __shfl_xor(bi, off, 32);
      if (ov > bv || (ov == bv && oi > bi)) { bv = ov; bi = oi; }
    }
    if (ci == bi && bi >= 0) --pos;        // exactly one lane owns bi
    if (lane == 0) out[r] = (bi < 0) ? 0 : bi;
  }
}

// ---------------- per-token solver kernels (128 thr = 4 waves) ----------------

__global__ void k_init(const float* __restrict__ q, const float* __restrict__ memory,
                       const int* __restrict__ kset, float* __restrict__ P,
                       float* __restrict__ Y) {
  __shared__ float qs[D_DIM];
  __shared__ float s[SHORT_K];
  __shared__ float p[SHORT_K];
  __shared__ int rows[SHORT_K];
  int t = blockIdx.x, tid = threadIdx.x;
  int lane = tid & 31, w = tid >> 5;
  if (tid < SHORT_K) rows[tid] = kset[t * SHORT_K + tid];
  for (int d = tid; d < D_DIM; d += 128) qs[d] = q[(size_t)t * D_DIM + d];
  __syncthreads();
  for (int k = w * 8; k < w * 8 + 8; ++k) {
    const float* mr = memory + (size_t)rows[k] * D_DIM;
    float acc = 0.f;
    for (int d = lane; d < D_DIM; d += 32) acc += qs[d] * mr[d];
    for (int off = 16; off; off >>= 1) acc += __shfl_xor(acc, off, 32);
    if (lane == 0) s[k] = acc * 0.03125f;   // 1/sqrt(1024)
  }
  __syncthreads();
  if (w == 0) {
    float v = s[lane];
    float mx = v;
    for (int off = 16; off; off >>= 1) { float o = __shfl_xor(mx, off, 32); mx = o > mx ? o : mx; }
    float e = expf(v - mx);
    float sum = e;
    for (int off = 16; off; off >>= 1) sum += __shfl_xor(sum, off, 32);
    float pv = e / sum;
    p[lane] = pv;
    P[(size_t)t * SHORT_K + lane] = pv;
  }
  __syncthreads();
  for (int d = tid; d < D_DIM; d += 128) {
    float acc = 0.f;
    #pragma unroll 8
    for (int k = 0; k < SHORT_K; ++k) acc += p[k] * memory[(size_t)rows[k] * D_DIM + d];
    Y[(size_t)t * D_DIM + d] = acc;
  }
}

__global__ void k_lam(float* __restrict__ Lam, const float* __restrict__ AY,
                      const float* __restrict__ X, float tau, long n) {
  long i = (long)blockIdx.x * blockDim.x + threadIdx.x;
  long stride = (long)gridDim.x * blockDim.x;
  for (; i < n; i += stride) Lam[i] += tau * (AY[i] - X[i]);
}

__global__ void k_yup(float* __restrict__ Y, const float* __restrict__ P,
                      const float* __restrict__ memory, const int* __restrict__ kset,
                      const float* __restrict__ LamA, float tau) {
  __shared__ float p[SHORT_K];
  __shared__ int rows[SHORT_K];
  int t = blockIdx.x, tid = threadIdx.x;
  if (tid < SHORT_K) {
    p[tid] = P[(size_t)t * SHORT_K + tid];
    rows[tid] = kset[t * SHORT_K + tid];
  }
  __syncthreads();
  for (int d = tid; d < D_DIM; d += 128) {
    float pm = 0.f;
    #pragma unroll 8
    for (int k = 0; k < SHORT_K; ++k) pm += p[k] * memory[(size_t)rows[k] * D_DIM + d];
    size_t idx = (size_t)t * D_DIM + d;
    float y = Y[idx];
    Y[idx] = y - tau * (y - pm + LamA[idx]);
  }
}

__global__ void k_gup(float* __restrict__ P, const float* __restrict__ Y,
                      const float* __restrict__ memory, const int* __restrict__ kset,
                      float beta) {
  __shared__ float ys[D_DIM];
  __shared__ float g[SHORT_K];
  __shared__ int rows[SHORT_K];
  int t = blockIdx.x, tid = threadIdx.x, lane = tid & 31, w = tid >> 5;
  if (tid < SHORT_K) rows[tid] = kset[t * SHORT_K + tid];
  for (int d = tid; d < D_DIM; d += 128) ys[d] = Y[(size_t)t * D_DIM + d];
  __syncthreads();
  for (int k = w * 8; k < w * 8 + 8; ++k) {
    const float* mr = memory + (size_t)rows[k] * D_DIM;
    float acc = 0.f;
    for (int d = lane; d < D_DIM; d += 32) acc += ys[d] * mr[d];
    for (int off = 16; off; off >>= 1) acc += __shfl_xor(acc, off, 32);
    if (lane == 0) g[k] = acc;
  }
  __syncthreads();
  if (w == 0) {
    float lg = logf(P[(size_t)t * SHORT_K + lane] + 1e-9f) + beta * g[lane];
    float mx = lg;
    for (int off = 16; off; off >>= 1) { float o = __shfl_xor(mx, off, 32); mx = o > mx ? o : mx; }
    float e = expf(lg - mx), sum = e;
    for (int off = 16; off; off >>= 1) sum += __shfl_xor(sum, off, 32);
    P[(size_t)t * SHORT_K + lane] = e / sum;
  }
}

// ---------------- driver ----------------

extern "C" void kernel_launch(void* const* d_in, const int* in_sizes, int n_in,
                              void* d_out, int out_size, void* d_ws, size_t ws_size,
                              hipStream_t stream) {
  const int*   tokens    = (const int*)d_in[0];
  const float* embed     = (const float*)d_in[1];
  const float* memory    = (const float*)d_in[2];
  const float* scout_W   = (const float*)d_in[3];
  const float* field_W   = (const float*)d_in[4];
  const float* readout_W = (const float*)d_in[5];
  const float* readout_b = (const float*)d_in[6];
  float* out = (float*)d_out;

  char* wp = (char*)d_ws;
  auto alloc = [&](size_t bytes) -> char* {
    char* r = wp; wp += (bytes + 255) & ~(size_t)255; return r;
  };
  unsigned short* memB = (unsigned short*)alloc((size_t)KMEM * D_DIM * 2);
  float*          Ef   = (float*)alloc((size_t)NTOK * D_DIM * 4);
  unsigned short* Eb   = (unsigned short*)alloc((size_t)NTOK * D_DIM * 2);
  unsigned short* SWT  = (unsigned short*)alloc((size_t)D_DIM * D_DIM * 2);
  unsigned short* RWT  = (unsigned short*)alloc((size_t)VOCAB * D_DIM * 2);
  float*          Q    = (float*)alloc((size_t)NTOK * D_DIM * 4);
  float*          SC   = (float*)alloc((size_t)CHUNK_M * KMEM * 4);
  int*            KSET = (int*)alloc((size_t)NTOK * SHORT_K * 4);
  float*          P    = (float*)alloc((size_t)NTOK * SHORT_K * 4);
  float*          Y    = (float*)alloc((size_t)NTOK * D_DIM * 4);
  float*          LAM  = (float*)alloc((size_t)NTOK * D_DIM * 4);
  float*          AY   = (float*)alloc((size_t)NTOK * D_DIM * 4);
  float*          LAMA = (float*)alloc((size_t)NTOK * D_DIM * 4);
  float*          AM   = (float*)alloc((size_t)D_DIM * D_DIM * 4);
  float*          ATM  = (float*)alloc((size_t)D_DIM * D_DIM * 4);
  unsigned short* YB   = (unsigned short*)alloc((size_t)NTOK * D_DIM * 2);

  // prep
  k_cvt_bf16<<<2048, 256, 0, stream>>>(memory, memB, (long)KMEM * D_DIM);
  k_transpose_bf16<<<dim3(D_DIM / 16, D_DIM / 16), dim3(16, 16), 0, stream>>>(scout_W, SWT, D_DIM, D_DIM);
  k_transpose_bf16<<<dim3(VOCAB / 16, D_DIM / 16), dim3(16, 16), 0, stream>>>(readout_W, RWT, D_DIM, VOCAB);
  k_embed_gather<<<NTOK, 256, 0, stream>>>(tokens, embed, Ef, Eb);
  k_band<<<(D_DIM * D_DIM + 255) / 256, 256, 0, stream>>>(field_W, AM, ATM);
  k_zero<<<2048, 256, 0, stream>>>(LAM, (long)NTOK * D_DIM);

  // q = E @ scout_W   (NT with pre-transposed scout)
  k_gemm_nt_bf16<<<dim3(D_DIM / 64, NTOK / 64), 256, 0, stream>>>(Eb, SWT, Q, nullptr, NTOK, D_DIM, D_DIM);

  // scores + exact top-32, chunked over tokens to bound scratch
  for (int c = 0; c < NTOK / CHUNK_M; ++c) {
    k_gemm_nt_bf16<<<dim3(KMEM / 64, CHUNK_M / 64), 256, 0, stream>>>(
        Eb + (size_t)c * CHUNK_M * D_DIM, memB, SC, nullptr, CHUNK_M, KMEM, D_DIM);
    k_topk<<<CHUNK_M, 32, 0, stream>>>(SC, KSET, c * CHUNK_M);
  }

  // P0, Y0
  k_init<<<NTOK, 128, 0, stream>>>(Q, memory, KSET, P, Y);

  // 6 dual-ascent iterations; banded D x D matmuls in f32 WMMA
  const float betas[6] = {1.0f, 1.6f, 2.2f, 2.8f, 3.4f, 4.0f};
  const float taus[6]  = {0.5f, 0.42f, 0.34f, 0.26f, 0.18f, 0.10f};
  for (int it = 0; it < 6; ++it) {
    k_gemm_nt_f32<<<dim3(D_DIM / 32, NTOK / 64), 256, 0, stream>>>(Y, AM, AY, NTOK, D_DIM, D_DIM, 1);
    k_lam<<<2048, 256, 0, stream>>>(LAM, AY, Ef, taus[it], (long)NTOK * D_DIM);
    k_gemm_nt_f32<<<dim3(D_DIM / 32, NTOK / 64), 256, 0, stream>>>(LAM, ATM, LAMA, NTOK, D_DIM, D_DIM, 1);
    k_yup<<<NTOK, 128, 0, stream>>>(Y, P, memory, KSET, LAMA, taus[it]);
    k_gup<<<NTOK, 128, 0, stream>>>(P, Y, memory, KSET, betas[it]);
  }

  // logits = Y @ readout_W + b
  k_cvt_bf16<<<2048, 256, 0, stream>>>(Y, YB, (long)NTOK * D_DIM);
  k_gemm_nt_bf16<<<dim3(VOCAB / 64, NTOK / 64), 256, 0, stream>>>(YB, RWT, out, readout_b, NTOK, VOCAB, D_DIM);
}
